// SNNModel_67611375174158
// MI455X (gfx1250) — matrix operand order repaired and checked by reading, not compiled, not run on hardware
//
#include <hip/hip_runtime.h>
#include <hip/hip_bf16.h>
#include <stdint.h>

#define B_    128
#define T_    128
#define NIN   2312     // 2*34*34
#define NHID  2048
#define NOUT  10
#define MSZ   16
#define ALPHA_ 0.001f

typedef __attribute__((ext_vector_type(16))) _Float16 v16h;
typedef __attribute__((ext_vector_type(8)))  float    v8f;

#ifndef __has_builtin
#define __has_builtin(x) 0
#endif

#if __has_builtin(__builtin_amdgcn_global_load_async_to_lds_b128)
#define HAVE_ASYNC 1
#else
#define HAVE_ASYNC 0
#endif

// The async builtin's params are pointers to a 16B generic vector type:
//   (V4i addrspace(1)*, V4i addrspace(3)*, imm offset, imm cpol)
typedef int vi4 __attribute__((vector_size(16)));
typedef __attribute__((address_space(1))) vi4 GV4;
typedef __attribute__((address_space(3))) vi4 LV4;

__device__ __forceinline__ void ld_async_b128(const int* g, int* l) {
#if HAVE_ASYNC
  __builtin_amdgcn_global_load_async_to_lds_b128((GV4*)const_cast<int*>(g),
                                                 (LV4*)l, 0, 0);
#else
  ((int4*)l)[0] = ((const int4*)g)[0];
#endif
}

__device__ __forceinline__ void wait_async0() {
#if HAVE_ASYNC
#if __has_builtin(__builtin_amdgcn_s_wait_asynccnt)
  __builtin_amdgcn_s_wait_asynccnt(0);
#else
  asm volatile("s_wait_asynccnt 0" ::: "memory");
#endif
#endif
}

// ---------------------------------------------------------------------------
// Kernel 0: zero the has[] flags in workspace (ws is poisoned, must init).
// ---------------------------------------------------------------------------
__global__ void k_init(int* __restrict__ has) {
  has[threadIdx.x] = 0;
}

// ---------------------------------------------------------------------------
// Kernel 1: stream frames (151.6 MB, the HBM-bound part).
// One wave per (b,t) row: async-copy 2312 ints into LDS, then XOR-reduce the
// active input ids and compute the any-active flag. Butterfly reduce with
// wave32 shuffles; lane 0 writes acc (transposed [t][b] for coalesced reads
// in the scan kernel) and atomically ORs has[t].
// ---------------------------------------------------------------------------
__global__ __launch_bounds__(128) void k_stream(
    const int* __restrict__ frames, const int* __restrict__ in_ids,
    unsigned* __restrict__ accA, int* __restrict__ has) {
  __shared__ int s_ids[NIN];
  __shared__ int s_fr[4 * NIN];
  const int tid  = threadIdx.x;
  const int wave = tid >> 5;
  const int lane = tid & 31;
  const int r    = blockIdx.x * 4 + wave;   // flat (b,t): r = b*T + t
  const int* g   = frames + (size_t)r * NIN;
  int* lf        = &s_fr[wave * NIN];

  // Issue all async copies for this wave's row (578 x b128 = 9248 B).
  for (int vi = lane; vi < NIN / 4; vi += 32)
    ld_async_b128(g + vi * 4, lf + vi * 4);

  // Stage the id table cooperatively while the async copies fly.
  for (int i = tid; i < NIN; i += 128) s_ids[i] = in_ids[i];

  wait_async0();
  __syncthreads();

  unsigned acc = 0u;
  int any = 0;
  for (int vi = lane; vi < NIN / 4; vi += 32) {
    int4 v = ((const int4*)lf)[vi];
    int j = vi * 4;
    if (v.x > 0) { acc ^= (unsigned)s_ids[j + 0]; any = 1; }
    if (v.y > 0) { acc ^= (unsigned)s_ids[j + 1]; any = 1; }
    if (v.z > 0) { acc ^= (unsigned)s_ids[j + 2]; any = 1; }
    if (v.w > 0) { acc ^= (unsigned)s_ids[j + 3]; any = 1; }
  }
#pragma unroll
  for (int off = 16; off >= 1; off >>= 1) {
    acc ^= __shfl_xor(acc, off, 32);
    any |= __shfl_xor(any, off, 32);
  }
  if (lane == 0) {
    const int b = r >> 7, t = r & (T_ - 1);
    accA[t * B_ + b] = acc;
    if (any) atomicOr(&has[t], 1);
  }
}

// ---------------------------------------------------------------------------
// Kernel 2: the sequential scan. One block, thread b = batch element.
// Prologue builds X1[m] = XOR of neuron_id0[h] over h active at mem_phys=m
// (hidden activations depend on mp0 only through this 16-entry table).
// Then run the 4-bit mp0/mp1 XOR recurrence over t, histogram mp values in
// LDS, and emit d0/d1 directly from the histograms.
// ---------------------------------------------------------------------------
__global__ __launch_bounds__(128) void k_scan(
    const unsigned* __restrict__ accA, const int* __restrict__ has,
    const float* __restrict__ v_th0, const int* __restrict__ nid0,
    const float* __restrict__ mm0, const float* __restrict__ v_th1,
    const float* __restrict__ mm1, int* __restrict__ mp1w,
    float* __restrict__ d0_out, float* __restrict__ d1_out) {
  __shared__ unsigned X1[MSZ];
  __shared__ int c0[MSZ], c1[MSZ];
  const int tid = threadIdx.x;
  if (tid < MSZ) { X1[tid] = 0u; c0[tid] = 0; c1[tid] = 0; }
  __syncthreads();

  unsigned xl[MSZ];
#pragma unroll
  for (int m = 0; m < MSZ; ++m) xl[m] = 0u;
  for (int h = tid; h < NHID; h += 128) {
    const float vt = v_th0[h];
    const unsigned id = (unsigned)nid0[h];
#pragma unroll
    for (int m = 0; m < MSZ; ++m)
      if (mm0[h * MSZ + m] >= vt) xl[m] ^= id;
  }
#pragma unroll
  for (int m = 0; m < MSZ; ++m)
    if (xl[m]) atomicXor(&X1[m], xl[m]);
  __syncthreads();

  const int b = tid;
  int t_last = 0, mp0 = 0, mp1 = 0;
  for (int t = 0; t < T_; ++t) {
    if (has[t]) {
      const int dt = (t - t_last) & 15;
      const unsigned a0 = accA[t * B_ + b];
      mp0 = (mp0 ^ dt ^ (int)(a0 & 15u)) & 15;
      const unsigned ax = X1[mp0];
      mp1 = (mp1 ^ dt ^ (int)(ax & 15u)) & 15;
      atomicAdd(&c0[mp0], 1);
      atomicAdd(&c1[mp1], 1);
      mp1w[t * B_ + b] = mp1;
      t_last = t;
    } else {
      mp1w[t * B_ + b] = MSZ;  // sentinel -> all-zero one-hot -> zero spikes
    }
  }
  __syncthreads();

  // d[h,m] = ALPHA*(m - mem_map[h,m]) * count[m]  (scatter collapsed to hist)
  for (int h = tid; h < NHID; h += 128) {
#pragma unroll
    for (int m = 0; m < MSZ; ++m)
      d0_out[h * MSZ + m] = ALPHA_ * ((float)m - mm0[h * MSZ + m]) * (float)c0[m];
  }
  if (tid < NOUT) {
#pragma unroll
    for (int m = 0; m < MSZ; ++m)
      d1_out[tid * MSZ + m] = ALPHA_ * ((float)m - mm1[tid * MSZ + m]) * (float)c1[m];
  }
}

// ---------------------------------------------------------------------------
// Kernel 3: spike raster via WMMA: out[bt, o] = onehot(mp1[bt]) @ S[m, o]
// where S[m,o] = (mem_map1[o,m] >= v_th1[o]). {0,1} operands are exact in
// f16, so v_wmma_f32_16x16x32_f16 reproduces the reference bit-exactly.
// One wave per 16-row tile: 1024 WMMA ops total.
// A layout (16x32 f16): lanes 0-15 hold K=0..7 in elems 0..7 (M=lane),
//                       lanes 16-31 hold K=8..15 in elems 0..7; K>=16 zero.
// B layout (32x16 f16): lanes 0-15 hold K=0..7 in elems 0..7 (N=lane),
//                       lanes 16-31 hold K=8..15 in elems 0..7; K>=16 zero.
// ---------------------------------------------------------------------------
__global__ __launch_bounds__(256) void k_raster(
    const int* __restrict__ mp1w, const float* __restrict__ mm1,
    const float* __restrict__ v_th1, float* __restrict__ spikes) {
  const int lane  = threadIdx.x & 31;
  const int wave  = threadIdx.x >> 5;
  const int tile  = blockIdx.x * 8 + wave;  // 1024 tiles of 16 rows
  const int rbase = tile * 16;
  const _Float16 one  = (_Float16)1.0f;
  const _Float16 zero = (_Float16)0.0f;

  // A: one-hot of mp1 for row (b*T + t) = rbase + (lane & 15)
  const int row = rbase + (lane & 15);
  const int bb  = row >> 7;
  const int tt  = row & (T_ - 1);
  const int mp  = mp1w[tt * B_ + bb];       // 0..15, or 16 (no-spike sentinel)
  const int ks  = (lane < 16) ? 0 : 8;
  v16h a;
#pragma unroll
  for (int e = 0; e < 16; ++e) a[e] = zero;
#pragma unroll
  for (int e = 0; e < 8; ++e) a[e] = (mp == ks + e) ? one : zero;

  // B: spike table column o = lane & 15 (padded to 16 cols, K=16..31 zero)
  const int o  = lane & 15;
  const int oc = (o < NOUT) ? o : 0;
  const float vt = v_th1[oc];
  v16h bm;
#pragma unroll
  for (int e = 0; e < 16; ++e) {
    const bool v = (e < 8) && (o < NOUT) && (mm1[oc * MSZ + ks + e] >= vt);
    bm[e] = v ? one : zero;
  }

  v8f c;
#pragma unroll
  for (int i = 0; i < 8; ++i) c[i] = 0.0f;
  c = __builtin_amdgcn_wmma_f32_16x16x32_f16(false, a, false, bm,
                                             (short)0, c, false, false);

  // D layout: VGPR r holds M=r (lanes 0-15) / M=8+r (lanes 16-31), N=lane&15
  const int M0 = (lane < 16) ? 0 : 8;
#pragma unroll
  for (int rr = 0; rr < 8; ++rr) {
    const int orow = rbase + M0 + rr;
    if (o < NOUT) spikes[(size_t)orow * NOUT + o] = c[rr];
  }
}

// ---------------------------------------------------------------------------
extern "C" void kernel_launch(void* const* d_in, const int* in_sizes, int n_in,
                              void* d_out, int out_size, void* d_ws, size_t ws_size,
                              hipStream_t stream) {
  const int*   frames = (const int*)d_in[0];
  const int*   in_ids = (const int*)d_in[1];
  const float* v_th0  = (const float*)d_in[3];
  const int*   nid0   = (const int*)d_in[4];
  const float* mm0    = (const float*)d_in[5];
  const float* v_th1  = (const float*)d_in[8];
  const float* mm1    = (const float*)d_in[10];

  // workspace layout: has[T] | accA[T*B] | mp1w[T*B]  (~132 KB)
  int*      has  = (int*)d_ws;
  unsigned* accA = (unsigned*)d_ws + T_;
  int*      mp1w = (int*)d_ws + T_ + T_ * B_;

  // output layout: spikes[B*T*10] | d0[2048*16] | d1[10*16]
  float* spikes = (float*)d_out;
  float* d0     = spikes + (size_t)B_ * T_ * NOUT;
  float* d1     = d0 + NHID * MSZ;

  k_init  <<<1, T_, 0, stream>>>(has);
  k_stream<<<(B_ * T_) / 4, 128, 0, stream>>>(frames, in_ids, accA, has);
  k_scan  <<<1, B_, 0, stream>>>(accA, has, v_th0, nid0, mm0, v_th1, mm1,
                                 mp1w, d0, d1);
  k_raster<<<(B_ * T_) / (16 * 8), 256, 0, stream>>>(mp1w, mm1, v_th1, spikes);
}
